// GT_SA_58660663329241
// MI455X (gfx1250) — matrix operand order repaired and checked by reading, not compiled, not run on hardware
//
#include <hip/hip_runtime.h>

typedef __attribute__((ext_vector_type(16))) _Float16 v16h;
typedef __attribute__((ext_vector_type(8)))  _Float16 v8h;
typedef __attribute__((ext_vector_type(4)))  _Float16 v4h;
typedef __attribute__((ext_vector_type(8)))  float    v8f;

#define NN  4096
#define IND 512
#define HH  8
#define DD  64
#define HD  512   // H * ATT_DIM == H * OUT_DIM

__device__ __forceinline__ float fast_rcp(float x) {
  return __builtin_amdgcn_rcpf(x);   // v_rcp_f32, avoids IEEE div expansion
}

__device__ __forceinline__ v8f wmma16(v16h a, v16h b, v8f c) {
  // v_wmma_f32_16x16x32_f16  (neg_a, A, neg_b, B, c_mod, C, reuse_a, reuse_b)
  return __builtin_amdgcn_wmma_f32_16x16x32_f16(false, a, false, b, (short)0, c, false, false);
}

// Build a 16-half fragment from two contiguous 8-half (16B) LDS chunks.
__device__ __forceinline__ v16h ld_frag(const _Float16* p0, const _Float16* p1) {
  v8h lo = *(const v8h*)p0;
  v8h hi = *(const v8h*)p1;
  v16h r;
#pragma unroll
  for (int i = 0; i < 8; ++i) { r[i] = lo[i]; r[i + 8] = hi[i]; }
  return r;
}

// ---------------------------------------------------------------------------
// Precompute E[n,m] = exp(0.1*adj[n,m] - 0.5) in f16 (head-independent factor
// of the post-sigmoid softmax numerator; reused by all 8 heads from L2).
// ---------------------------------------------------------------------------
__global__ __launch_bounds__(256)
void prep_adj(const float* __restrict__ adj, _Float16* __restrict__ E) {
  size_t i = ((size_t)blockIdx.x * 256 + threadIdx.x) * 4;
  float4 a = *(const float4*)&adj[i];
  v4h e;
  e[0] = (_Float16)__expf(0.1f * a.x - 0.5f);
  e[1] = (_Float16)__expf(0.1f * a.y - 0.5f);
  e[2] = (_Float16)__expf(0.1f * a.z - 0.5f);
  e[3] = (_Float16)__expf(0.1f * a.w - 0.5f);
  *(v4h*)&E[i] = e;
}

// ---------------------------------------------------------------------------
// Generic tiled WMMA GEMM: C[M x Nc] = ( A[M x Kd] @ W[Kd x Nc] + bias ) * outScale
// Block = 128 threads (4 waves, 2x2 wave grid), tile 64x64, BK = 32.
// A_HALF: A is f16 (else f32, converted on load). OUT_HALF: store f16 (else f32).
// ---------------------------------------------------------------------------
template <bool A_HALF, bool OUT_HALF>
__global__ __launch_bounds__(128)
void gemm_wmma(const void* __restrict__ Aptr, const float* __restrict__ W,
               const float* __restrict__ bias, void* __restrict__ Cptr,
               int M, int Kd, int Nc, float outScale) {
  __shared__ __align__(16) _Float16 As[64 * 32];  // [m][k]
  __shared__ __align__(16) _Float16 Bs[64 * 32];  // [n][k] (transposed)
  const int tid  = threadIdx.x;
  const int lane = tid & 31, wave = tid >> 5;
  const int wm = (wave >> 1) * 32, wn = (wave & 1) * 32;
  const int m0 = blockIdx.y * 64, n0 = blockIdx.x * 64;
  const int lr = lane & 15;
  const int la = (lane >> 4) * 8;    // A-fragment K offset
  const int lb = (lane >> 4) * 8;    // B-fragment K offset
  const int rhalf = (lane >> 4) * 8; // C/D row offset

  v8f acc[2][2] = {};

  for (int kt = 0; kt < Kd; kt += 32) {
    // ---- stage A tile (64x32, fp32->f16 if needed) ----
#pragma unroll
    for (int i = 0; i < 16; ++i) {
      int idx = tid + i * 128;               // 0..2047
      int r = idx >> 5, c = idx & 31;
      float av;
      if (A_HALF) av = (float)((const _Float16*)Aptr)[(size_t)(m0 + r) * Kd + kt + c];
      else        av = ((const float*)Aptr)[(size_t)(m0 + r) * Kd + kt + c];
      As[r * 32 + c] = (_Float16)av;
    }
    // ---- stage B tile transposed (W is [Kd][Nc] row-major) ----
#pragma unroll
    for (int i = 0; i < 16; ++i) {
      int idx = tid + i * 128;
      int k = idx >> 6, n = idx & 63;        // coalesced along n
      Bs[n * 32 + k] = (_Float16)W[(size_t)(kt + k) * Nc + n0 + n];
    }
    __syncthreads();

    v16h afr[2], bfr[2];
#pragma unroll
    for (int t = 0; t < 2; ++t) {
      int row = wm + t * 16 + lr;
      afr[t] = ld_frag(&As[row * 32 + la], &As[row * 32 + 16 + la]);
      int col = wn + t * 16 + lr;
      bfr[t] = ld_frag(&Bs[col * 32 + lb], &Bs[col * 32 + lb + 16]);
    }
#pragma unroll
    for (int ta = 0; ta < 2; ++ta)
#pragma unroll
      for (int tb = 0; tb < 2; ++tb)
        acc[ta][tb] = wmma16(afr[ta], bfr[tb], acc[ta][tb]);
    __syncthreads();
  }

  // ---- epilogue: bias + scale, store ----
#pragma unroll
  for (int ta = 0; ta < 2; ++ta)
#pragma unroll
    for (int tb = 0; tb < 2; ++tb)
#pragma unroll
      for (int r = 0; r < 8; ++r) {
        int row = m0 + wm + ta * 16 + r + rhalf;
        int col = n0 + wn + tb * 16 + lr;
        float val = (acc[ta][tb][r] + bias[col]) * outScale;
        if (OUT_HALF) ((_Float16*)Cptr)[(size_t)row * Nc + col] = (_Float16)val;
        else          ((float*)Cptr)[(size_t)row * Nc + col]    = val;
      }
}

// ---------------------------------------------------------------------------
// Fused attention (flash-style, no running max needed: post-sigmoid scores are
// bounded in (-0.5, 0.6)). One block = (head h, 64 query rows), 4 waves.
// Each wave owns 16 query rows; loops over 64-key blocks:
//   S = Qblk @ Kblk^T (WMMA) -> p = exp(sigmoid(S)) * E[n,m]
//   rowsum += p ; O += P @ Vblk (WMMA). Final: O *= rcp(rowsum).
// ---------------------------------------------------------------------------
__global__ __launch_bounds__(128)
void attn_wmma(const _Float16* __restrict__ q16, const _Float16* __restrict__ k16,
               const _Float16* __restrict__ v16m, const _Float16* __restrict__ E16,
               _Float16* __restrict__ o16) {
  __shared__ __align__(16) _Float16 Qs[64 * 64];  // [qrow][d]
  __shared__ __align__(16) _Float16 Ks[64 * 64];  // [key][d]
  __shared__ __align__(16) _Float16 Vs[64 * 64];  // transposed: [e][key]
  __shared__ __align__(16) _Float16 Ps[64 * 64];  // [qrow][key]
  const int tid  = threadIdx.x;
  const int lane = tid & 31, wave = tid >> 5;
  const int h = blockIdx.y, r0 = blockIdx.x * 64;
  const int lr = lane & 15;
  const int la = (lane >> 4) * 8;
  const int lb = (lane >> 4) * 8;
  const int rhalf = (lane >> 4) * 8;

  // stage Q block (vectorized 16B chunks)
#pragma unroll
  for (int i = 0; i < 4; ++i) {
    int c = tid + i * 128;                  // 512 chunks of 8 halfs
    int row = c >> 3, off = (c & 7) * 8;
    *(v8h*)&Qs[row * 64 + off] =
        *(const v8h*)&q16[(size_t)(r0 + row) * HD + h * DD + off];
  }

  v8f  accO[4] = {};
  float rs[8]  = {};

  for (int kb = 0; kb < NN; kb += 64) {
    // stage K (row-major) and V (transposed) blocks
#pragma unroll
    for (int i = 0; i < 4; ++i) {
      int c = tid + i * 128;
      int row = c >> 3, off = (c & 7) * 8;
      *(v8h*)&Ks[row * 64 + off] =
          *(const v8h*)&k16[(size_t)(kb + row) * HD + h * DD + off];
      v8h vv = *(const v8h*)&v16m[(size_t)(kb + row) * HD + h * DD + off];
#pragma unroll
      for (int j = 0; j < 8; ++j) Vs[(off + j) * 64 + row] = vv[j];
    }
    __syncthreads();

    if (kb + 64 < NN) {  // gfx1250 global_prefetch_b8 for next tiles
      __builtin_prefetch(&k16[(size_t)(kb + 64) * HD + h * DD], 0, 0);
      __builtin_prefetch(&v16m[(size_t)(kb + 64) * HD + h * DD], 0, 0);
    }

    // ---- S = Q @ K^T over d=64 (two K=32 WMMA steps, 4 col tiles) ----
    v8f accS[4] = {};
#pragma unroll
    for (int ks = 0; ks < 64; ks += 32) {
      int row = wave * 16 + lr;
      v16h af = ld_frag(&Qs[row * 64 + ks + la], &Qs[row * 64 + ks + 16 + la]);
#pragma unroll
      for (int ct = 0; ct < 4; ++ct) {
        int key = ct * 16 + lr;
        v16h bf = ld_frag(&Ks[key * 64 + ks + lb], &Ks[key * 64 + ks + lb + 16]);
        accS[ct] = wmma16(af, bf, accS[ct]);
      }
    }

    // ---- elementwise: p = exp(sigmoid(s)) * E[n,m]; rowsum; P -> LDS ----
#pragma unroll
    for (int ct = 0; ct < 4; ++ct) {
      int keyg = kb + ct * 16 + lr;
#pragma unroll
      for (int r = 0; r < 8; ++r) {
        int qrow = wave * 16 + r + rhalf;
        float s  = accS[ct][r];                     // q-scale folded into q16
        float sg = fast_rcp(1.0f + __expf(-s));     // v_rcp_f32 sigmoid
        float p  = __expf(sg) *
                   (float)E16[(size_t)(r0 + qrow) * NN + keyg];
        rs[r] += p;
        Ps[qrow * 64 + ct * 16 + lr] = (_Float16)p;
      }
    }
    __syncthreads();

    // ---- O += P @ V (K dim = 64 keys) ----
#pragma unroll
    for (int ks = 0; ks < 64; ks += 32) {
      int row = wave * 16 + lr;
      v16h af = ld_frag(&Ps[row * 64 + ks + la], &Ps[row * 64 + ks + 16 + la]);
#pragma unroll
      for (int ct = 0; ct < 4; ++ct) {
        int e = ct * 16 + lr;
        v16h bf = ld_frag(&Vs[e * 64 + ks + lb], &Vs[e * 64 + ks + lb + 16]);
        accO[ct] = wmma16(af, bf, accO[ct]);
      }
    }
    __syncthreads();
  }

  // reduce rowsums across the 16 lanes of each half-wave (matches C layout)
#pragma unroll
  for (int r = 0; r < 8; ++r)
#pragma unroll
    for (int m = 1; m < 16; m <<= 1)
      rs[r] += __shfl_xor(rs[r], m, 32);

  // normalize (one v_rcp per row, multiply) and store f16 output [N][H*e]
#pragma unroll
  for (int r = 0; r < 8; ++r) {
    float inv = fast_rcp(rs[r]);
    int qrow = wave * 16 + r + rhalf;
#pragma unroll
    for (int ct = 0; ct < 4; ++ct) {
      int col = ct * 16 + lr;
      o16[(size_t)(r0 + qrow) * HD + h * DD + col] =
          (_Float16)(accO[ct][r] * inv);
    }
  }
}

// ---------------------------------------------------------------------------
extern "C" void kernel_launch(void* const* d_in, const int* in_sizes, int n_in,
                              void* d_out, int out_size, void* d_ws, size_t ws_size,
                              hipStream_t stream) {
  const float* Q   = (const float*)d_in[0];
  const float* K   = (const float*)d_in[1];
  const float* V   = (const float*)d_in[2];
  const float* adj = (const float*)d_in[3];
  const float* WQ  = (const float*)d_in[4];
  const float* bQ  = (const float*)d_in[5];
  const float* WK  = (const float*)d_in[6];
  const float* bK  = (const float*)d_in[7];
  const float* WV  = (const float*)d_in[8];
  const float* bV  = (const float*)d_in[9];
  const float* WO  = (const float*)d_in[10];
  const float* bO  = (const float*)d_in[11];

  _Float16* q16 = (_Float16*)d_ws;                  //  4 MB
  _Float16* k16 = q16 + (size_t)NN * HD;            //  4 MB
  _Float16* vw  = k16 + (size_t)NN * HD;            //  4 MB
  _Float16* o16 = vw  + (size_t)NN * HD;            //  4 MB
  _Float16* E16 = o16 + (size_t)NN * HD;            // 32 MB (L2-resident)
  (void)in_sizes; (void)n_in; (void)out_size; (void)ws_size;

  const float scale = 0.125f;  // 1/sqrt(ATT_DIM), folded into q16
  dim3 blk(128);

  // head-independent adj factor (bandwidth-bound, runs once)
  prep_adj<<<((size_t)NN * NN / 4) / 256, 256, 0, stream>>>(adj, E16);

  // QKV projections (f32 in -> f16 out)
  gemm_wmma<false, true><<<dim3(HD / 64, NN / 64), blk, 0, stream>>>(
      Q, WQ, bQ, q16, NN, IND, HD, scale);
  gemm_wmma<false, true><<<dim3(HD / 64, NN / 64), blk, 0, stream>>>(
      K, WK, bK, k16, NN, IND, HD, 1.0f);
  gemm_wmma<false, true><<<dim3(HD / 64, NN / 64), blk, 0, stream>>>(
      V, WV, bV, vw, NN, IND, HD, 1.0f);

  // fused sigmoid-softmax attention, all heads
  attn_wmma<<<dim3(NN / 64, HH), blk, 0, stream>>>(q16, k16, vw, E16, o16);

  // output projection (f16 in -> f32 out)
  gemm_wmma<true, false><<<dim3(DD / 64, NN / 64), blk, 0, stream>>>(
      o16, WO, bO, (float*)d_out, NN, HD, DD, 1.0f);
}
